// HANLayer_1425929143036
// MI455X (gfx1250) — matrix operand order repaired and checked by reading, not compiled, not run on hardware
//
#include <hip/hip_runtime.h>
#include <hip/hip_bf16.h>
#include <math.h>

#define N_NODES 50000
#define N_EDGES 600000
#define N_PATHS 3
#define DIM 128

typedef __attribute__((ext_vector_type(16))) __bf16 v16bf;
typedef __attribute__((ext_vector_type(8)))  __bf16 v8bf;
typedef __attribute__((ext_vector_type(8)))  float  v8f;

// ---------------- workspace layout (floats) ----------------
// zBuf   : [3][50000][128]         19,200,000
// degOut : [3][50000]                 150,000
// degIn  : [3][50000]                 150,000
// wAcc   : [4]   (at 19,500,000)
// beta   : [4]   (at 19,500,004)
#define Z_FLOATS   (N_PATHS * N_NODES * DIM)          // 19,200,000
#define DEGO_OFF   Z_FLOATS
#define DEGI_OFF   (Z_FLOATS + N_PATHS * N_NODES)
#define WACC_OFF   19500000
#define BETA_OFF   19500004
#define ZERO_F4    ((19500008) / 4)

// ---------------- K0: zero workspace ----------------
__global__ void __launch_bounds__(256) zero_kernel(float4* ws) {
    int i = blockIdx.x * 256 + threadIdx.x;
    if (i < ZERO_F4) ws[i] = make_float4(0.f, 0.f, 0.f, 0.f);
}

// ---------------- K1: degree counts ----------------
__global__ void __launch_bounds__(256) degree_kernel(const int* __restrict__ esrc,
                                                     const int* __restrict__ edst,
                                                     float* degOut, float* degIn) {
    int i = blockIdx.x * 256 + threadIdx.x;
    if (i >= N_PATHS * N_EDGES) return;
    int p = i / N_EDGES;
    unsafeAtomicAdd(degOut + p * N_NODES + esrc[i], 1.0f);
    unsafeAtomicAdd(degIn  + p * N_NODES + edst[i], 1.0f);
}

// ---------------- shared WMMA tile helper ----------------
__device__ __forceinline__ v8f wmma_tile_128(const __bf16* sA, const __bf16* sBt,
                                             int lane, int colTile) {
    const int M    = lane & 15;
    const int half = lane >> 4;
    const int Nc   = colTile * 16 + (lane & 15);
    v8f c = {};
#pragma unroll
    for (int kt = 0; kt < 4; ++kt) {
        // A fragment: 16-bit A 16x32 layout: lo-half lanes hold K 0..7 & 16..23 (+kt*32)
        int akb = kt * 32 + half * 8;
        v8bf al = *(const v8bf*)(sA + M * DIM + akb);
        v8bf ah = *(const v8bf*)(sA + M * DIM + akb + 16);
        v16bf a = __builtin_shufflevector(al, ah, 0,1,2,3,4,5,6,7,8,9,10,11,12,13,14,15);
        // B fragment: lane holds column Nc, same K order as A (lo lanes K 0..7 & 16..23, hi 8..15 & 24..31)
        int bkb = kt * 32 + half * 8;
        v8bf bl = *(const v8bf*)(sBt + Nc * DIM + bkb);
        v8bf bh = *(const v8bf*)(sBt + Nc * DIM + bkb + 16);
        v16bf b = __builtin_shufflevector(bl, bh, 0,1,2,3,4,5,6,7,8,9,10,11,12,13,14,15);
        c = __builtin_amdgcn_wmma_f32_16x16x32_bf16(false, a, false, b, (short)0, c,
                                                    false, false);
    }
    return c;
}

// ---------------- K3: xW = (h * rsqrt(deg_out)) @ W_gc[p] ----------------
__global__ void __launch_bounds__(256) gemm_xw_kernel(const float* __restrict__ h,
                                                      const float* __restrict__ Wgc,
                                                      const float* __restrict__ degOut,
                                                      float* __restrict__ xW, int p) {
    __shared__ __align__(16) __bf16 sWt[DIM * DIM];  // [n][k]
    __shared__ __align__(16) __bf16 sA[16 * DIM];    // [m][k]
    const int tid = threadIdx.x;
    const float* W = Wgc + p * DIM * DIM;            // [k][n]
    for (int i = tid; i < DIM * DIM; i += 256) {
        int k = i >> 7, n = i & 127;
        sWt[n * DIM + k] = (__bf16)W[i];
    }
    const int row0 = blockIdx.x * 16;
    for (int i = tid; i < 16 * DIM; i += 256) {
        int m = i >> 7;
        int nrow = row0 + m;
        float s = rsqrtf(fmaxf(degOut[p * N_NODES + nrow], 1.0f));
        sA[i] = (__bf16)(h[(size_t)nrow * DIM + (i & 127)] * s);
    }
    __syncthreads();
    const int wave = tid >> 5, lane = tid & 31;
    v8f c = wmma_tile_128(sA, sWt, lane, wave);
    const int half = lane >> 4;
    const int Nc = wave * 16 + (lane & 15);
#pragma unroll
    for (int r = 0; r < 8; ++r) {
        int m = r + half * 8;                        // C layout: VGPR r -> M=r (lo) / r+8 (hi)
        xW[(size_t)(row0 + m) * DIM + Nc] = c[r];
    }
}

// ---------------- K4: edge scatter z[p][dst] += xW[src] ----------------
__global__ void __launch_bounds__(256) scatter_kernel(const float* __restrict__ xW,
                                                      const int* __restrict__ esrc,
                                                      const int* __restrict__ edst,
                                                      float* __restrict__ zp) {
    const int wave = threadIdx.x >> 5, lane = threadIdx.x & 31;
    const int e = blockIdx.x * 8 + wave;
    if (e >= N_EDGES) return;
    const int s = esrc[e], d = edst[e];
    float4 v = ((const float4*)(xW + (size_t)s * DIM))[lane];
    float* zd = zp + (size_t)d * DIM + lane * 4;
    unsafeAtomicAdd(zd + 0, v.x);
    unsafeAtomicAdd(zd + 1, v.y);
    unsafeAtomicAdd(zd + 2, v.z);
    unsafeAtomicAdd(zd + 3, v.w);
}

// ---------------- K5: z = z * rsqrt(deg_in) + b_gc ----------------
__global__ void __launch_bounds__(256) scalebias_kernel(float4* __restrict__ z,
                                                        const float* __restrict__ degIn,
                                                        const float* __restrict__ bgc) {
    int i = blockIdx.x * 256 + threadIdx.x;              // float4 index
    if (i >= N_PATHS * N_NODES * 32) return;
    int p = i / (N_NODES * 32);
    int rem = i - p * (N_NODES * 32);
    int n = rem >> 5, c4 = rem & 31;
    float s = rsqrtf(fmaxf(degIn[p * N_NODES + n], 1.0f));
    float4 bb = ((const float4*)(bgc + p * DIM))[c4];
    float4 v = z[i];
    v.x = v.x * s + bb.x; v.y = v.y * s + bb.y;
    v.z = v.z * s + bb.z; v.w = v.w * s + bb.w;
    z[i] = v;
}

// ---------------- K6: wAcc[p] += sum_n tanh(z@w1+b1)@w2 ----------------
__global__ void __launch_bounds__(256) attn_kernel(const float* __restrict__ z,
                                                   const float* __restrict__ w1,
                                                   const float* __restrict__ b1,
                                                   const float* __restrict__ w2,
                                                   float* wAcc) {
    __shared__ __align__(16) __bf16 sWt[DIM * DIM];
    __shared__ __align__(16) __bf16 sA[16 * DIM];
    __shared__ float bsum;
    const int tid = threadIdx.x, p = blockIdx.y;
    for (int i = tid; i < DIM * DIM; i += 256) {
        int k = i >> 7, n = i & 127;
        sWt[n * DIM + k] = (__bf16)w1[i];
    }
    const float* zrow = z + (size_t)p * N_NODES * DIM + (size_t)blockIdx.x * 16 * DIM;
    for (int i = tid; i < 16 * DIM; i += 256) sA[i] = (__bf16)zrow[i];
    if (tid == 0) bsum = 0.f;
    __syncthreads();
    const int wave = tid >> 5, lane = tid & 31;
    v8f c = wmma_tile_128(sA, sWt, lane, wave);
    const int Nc = wave * 16 + (lane & 15);
    float b1c = b1[Nc], w2c = w2[Nc];
    float part = 0.f;
#pragma unroll
    for (int r = 0; r < 8; ++r) part += tanhf(c[r] + b1c);
    part *= w2c;                 // all rows of this block, this column -> just sum
    atomicAdd(&bsum, part);      // ds_add_f32
    __syncthreads();
    if (tid == 0) unsafeAtomicAdd(wAcc + p, bsum);
}

// ---------------- K7a: beta = softmax(wAcc / N) ----------------
__global__ void softmax3_kernel(const float* wAcc, float* beta) {
    if (threadIdx.x == 0) {
        float a = wAcc[0] / N_NODES, b = wAcc[1] / N_NODES, c = wAcc[2] / N_NODES;
        float m = fmaxf(a, fmaxf(b, c));
        float ea = __expf(a - m), eb = __expf(b - m), ec = __expf(c - m);
        float s = ea + eb + ec;
        beta[0] = ea / s; beta[1] = eb / s; beta[2] = ec / s;
    }
}

// ---------------- K7b: out = sum_p beta[p] * z[p] ----------------
__global__ void __launch_bounds__(256) combine_kernel(const float4* __restrict__ z,
                                                      const float* __restrict__ beta,
                                                      float4* __restrict__ out) {
    int i = blockIdx.x * 256 + threadIdx.x;              // float4 over N*32
    if (i >= N_NODES * 32) return;
    float b0 = beta[0], b1 = beta[1], b2 = beta[2];
    float4 r0 = z[i], r1 = z[i + N_NODES * 32], r2 = z[i + 2 * N_NODES * 32];
    float4 o;
    o.x = b0 * r0.x + b1 * r1.x + b2 * r2.x;
    o.y = b0 * r0.y + b1 * r1.y + b2 * r2.y;
    o.z = b0 * r0.z + b1 * r1.z + b2 * r2.z;
    o.w = b0 * r0.w + b1 * r1.w + b2 * r2.w;
    out[i] = o;
}

extern "C" void kernel_launch(void* const* d_in, const int* in_sizes, int n_in,
                              void* d_out, int out_size, void* d_ws, size_t ws_size,
                              hipStream_t stream) {
    const float* h    = (const float*)d_in[0];
    const float* Wgc  = (const float*)d_in[1];
    const float* bgc  = (const float*)d_in[2];
    const float* w1   = (const float*)d_in[3];
    const float* b1   = (const float*)d_in[4];
    const float* w2   = (const float*)d_in[5];
    const int*   esrc = (const int*)d_in[6];
    const int*   edst = (const int*)d_in[7];
    float* out = (float*)d_out;
    float* ws  = (float*)d_ws;

    float* zBuf   = ws;
    float* degOut = ws + DEGO_OFF;
    float* degIn  = ws + DEGI_OFF;
    float* wAcc   = ws + WACC_OFF;
    float* beta   = ws + BETA_OFF;

    // K0: zero z, degrees, wAcc, beta
    zero_kernel<<<(ZERO_F4 + 255) / 256, 256, 0, stream>>>((float4*)ws);
    // K1: degrees
    degree_kernel<<<(N_PATHS * N_EDGES + 255) / 256, 256, 0, stream>>>(esrc, edst,
                                                                       degOut, degIn);
    // Per path: GEMM into d_out (scratch), then edge scatter into z[p]
    for (int p = 0; p < N_PATHS; ++p) {
        gemm_xw_kernel<<<N_NODES / 16, 256, 0, stream>>>(h, Wgc, degOut, out, p);
        scatter_kernel<<<N_EDGES / 8, 256, 0, stream>>>(out, esrc + p * N_EDGES,
                                                        edst + p * N_EDGES,
                                                        zBuf + (size_t)p * N_NODES * DIM);
    }
    // K5: z = z * rsqrt(deg_in) + b
    scalebias_kernel<<<(N_PATHS * N_NODES * 32 + 255) / 256, 256, 0, stream>>>(
        (float4*)zBuf, degIn, bgc);
    // K6: semantic attention logits
    attn_kernel<<<dim3(N_NODES / 16, N_PATHS), 256, 0, stream>>>(zBuf, w1, b1, w2, wAcc);
    // K7a: softmax over 3 path logits
    softmax3_kernel<<<1, 32, 0, stream>>>(wAcc, beta);
    // K7b: weighted combine into d_out
    combine_kernel<<<(N_NODES * 32 + 255) / 256, 256, 0, stream>>>((const float4*)zBuf,
                                                                   beta, (float4*)out);
}
